// CryoformerDecoderLayer_18245021073737
// MI455X (gfx1250) — matrix operand order repaired and checked by reading, not compiled, not run on hardware
//
#include <hip/hip_runtime.h>

// ---------------------------------------------------------------------------
// CryoformerDecoderLayer for MI455X (gfx1250, wave32, WMMA).
// All GEMMs run through v_wmma_f32_16x16x32_bf16 (bf16 A/B, f32 accum).
// LayerNorm / softmax statistics stay in fp32.
// Global->LDS staging uses GLOBAL_LOAD_ASYNC_TO_LDS_B128 when the builtin
// is available (ASYNCcnt path), otherwise 128-bit load/store staging.
// ---------------------------------------------------------------------------

#define D_MODEL 512
#define NHEAD   8
#define DHEAD   64
#define FFDIM   2048
#define MSA_DIM 256
#define PAIR_DIM 128
#define NRES    512
#define BATCH   4
#define NDEN    4096
#define MROWS   (NRES * BATCH)   // 2048
#define DROWS   (NDEN * BATCH)   // 16384
#define NBH     (BATCH * NHEAD)  // 32

typedef __attribute__((ext_vector_type(16))) __bf16       v16bf;
typedef __attribute__((ext_vector_type(8)))  float        v8f;
typedef __attribute__((ext_vector_type(8)))  unsigned int v8u;

#if defined(__has_builtin)
#if __has_builtin(__builtin_amdgcn_global_load_async_to_lds_b128)
#define ASYNC_LDS 1
#endif
#if __has_builtin(__builtin_amdgcn_s_wait_asynccnt)
#define HAVE_WAIT_ASYNC_BUILTIN 1
#endif
#endif
#ifndef ASYNC_LDS
#define ASYNC_LDS 0
#endif
#ifndef HAVE_WAIT_ASYNC_BUILTIN
#define HAVE_WAIT_ASYNC_BUILTIN 0
#endif

#if ASYNC_LDS
// Builtin parameter type is a 16-byte int vector pointer (per hipcc
// diagnostic): global (AS1) source, LDS (AS3) destination.
typedef int v4i_gcc __attribute__((vector_size(16)));
typedef v4i_gcc __attribute__((address_space(1)))* v4i_g1p;
typedef v4i_gcc __attribute__((address_space(3)))* v4i_l3p;
__device__ __forceinline__ void async_copy_b128(const void* g, void* l) {
  __builtin_amdgcn_global_load_async_to_lds_b128((v4i_g1p)g, (v4i_l3p)l, 0, 0);
}
__device__ __forceinline__ void async_wait0() {
#if HAVE_WAIT_ASYNC_BUILTIN
  __builtin_amdgcn_s_wait_asynccnt(0);
#else
  asm volatile("s_wait_asynccnt 0x0" ::: "memory");
#endif
}
#endif

__device__ __forceinline__ unsigned short f2bf(float f) {
  unsigned int u = __builtin_bit_cast(unsigned int, f);
  u += 0x7FFFu + ((u >> 16) & 1u);   // round to nearest even
  return (unsigned short)(u >> 16);
}

// ---------------------------------------------------------------------------
// Generic batched GEMM: C[g] = alpha * A[g] (MxK) * B[g] (NxK)^T
//                              + bias[N] + beta * addC[g] (MxN), opt ReLU.
// A, B are bf16 (as ushort); C is fp32 or bf16 (template OUTBF).
// amode/bmode/cmode: 0 = linear batch stride, 1 = attention head-merge
//   (base = (g/H)*D + (g%H)*64 into a merged [rows, B*D] buffer).
// ---------------------------------------------------------------------------
struct GemmP {
  const unsigned short* A;
  const unsigned short* B;
  float*          Cf;
  unsigned short* Cb;
  const float*    bias;
  const float*    addC;
  long long sA, sB, sC, sAdd;
  int lda, ldb, ldc;
  int M, N, K;
  float alpha, beta;
  int relu;
  int amode, bmode, cmode;
};

#define BM 128
#define BN 64
#define BK 64
#define LDS_STR 72   // padded row stride (bf16 elems): 144B rows, 16B aligned

template <bool BIAS, bool ADDC, bool RELU, bool OUTBF>
__global__ __launch_bounds__(256) void gemm_bf16_wmma(GemmP p) {
  __shared__ __align__(16) unsigned short As[BM * LDS_STR];
  __shared__ __align__(16) unsigned short Bs[BN * LDS_STR];

  const int g = blockIdx.z;
  const long long mergeBase = (long long)((g >> 3) * D_MODEL + (g & 7) * DHEAD);
  const long long aBase = p.amode ? mergeBase : (long long)g * p.sA;
  const long long bBase = p.bmode ? mergeBase : (long long)g * p.sB;
  const long long cBase = p.cmode ? mergeBase : (long long)g * p.sC;

  const int tid  = threadIdx.x;
  const int lane = tid & 31;
  const int w    = tid >> 5;           // 8 waves
  const int m0   = (w >> 1) * 32;      // wave tile origin in block tile
  const int n0   = (w & 1) * 32;
  const int blockM = blockIdx.y * BM;
  const int blockN = blockIdx.x * BN;
  const int half = lane >> 4;          // lane group 0 / 1
  const int lm   = lane & 15;

  v8f acc[2][2] = {};

  for (int k0 = 0; k0 < p.K; k0 += BK) {
    __syncthreads();
    // Stage A tile: 128x64 bf16 = 1024 x 16B chunks, 4 per thread.
#pragma unroll
    for (int v = 0; v < 4; ++v) {
      int pi  = tid + v * 256;
      int row = pi >> 3;
      int kc  = (pi & 7) * 8;
      const unsigned short* src =
          p.A + aBase + (long long)(blockM + row) * p.lda + k0 + kc;
      unsigned short* dst = &As[row * LDS_STR + kc];
#if ASYNC_LDS
      async_copy_b128(src, dst);
#else
      *(int4*)dst = *(const int4*)src;
#endif
    }
    // Stage B tile (stored [n][k]): 64x64 bf16 = 512 chunks, 2 per thread.
#pragma unroll
    for (int v = 0; v < 2; ++v) {
      int pi = tid + v * 256;
      int n  = pi >> 3;
      int kc = (pi & 7) * 8;
      const unsigned short* src =
          p.B + bBase + (long long)(blockN + n) * p.ldb + k0 + kc;
      unsigned short* dst = &Bs[n * LDS_STR + kc];
#if ASYNC_LDS
      async_copy_b128(src, dst);
#else
      *(int4*)dst = *(const int4*)src;
#endif
    }
#if ASYNC_LDS
    async_wait0();
#endif
    __syncthreads();

    // Two 16x16x32 WMMA k-substeps per staged 64-wide tile.
#pragma unroll
    for (int ks = 0; ks < BK; ks += 32) {
      v16bf af[2], bfr[2];
#pragma unroll
      for (int mt = 0; mt < 2; ++mt) {
        // A 16x32: lanes 0-15 -> K=0..15; lanes 16-31 -> K=16..31
        int row = m0 + mt * 16 + lm;
        int kh  = ks + half * 16;
        v8u au;
#pragma unroll
        for (int i = 0; i < 8; ++i)
          au[i] = *(const unsigned int*)(&As[row * LDS_STR + kh + 2 * i]);
        af[mt] = __builtin_bit_cast(v16bf, au);
      }
#pragma unroll
      for (int nt = 0; nt < 2; ++nt) {
        // B 32x16: lanes 0-15 -> K=0..15; lanes 16-31 -> K=16..31
        int n  = n0 + nt * 16 + lm;
        int kb = ks + half * 16;
        v8u bu;
#pragma unroll
        for (int i = 0; i < 8; ++i)
          bu[i] = *(const unsigned int*)(&Bs[n * LDS_STR + kb + 2 * i]);
        bfr[nt] = __builtin_bit_cast(v16bf, bu);
      }
#pragma unroll
      for (int mt = 0; mt < 2; ++mt)
#pragma unroll
        for (int nt = 0; nt < 2; ++nt)
          acc[mt][nt] = __builtin_amdgcn_wmma_f32_16x16x32_bf16(
              false, af[mt], false, bfr[nt], (short)0, acc[mt][nt], false, false);
    }
  }

  // Epilogue: C/D 16x16 f32 layout -> VGPR r: M = r + half*8, N = lm.
#pragma unroll
  for (int mt = 0; mt < 2; ++mt) {
#pragma unroll
    for (int nt = 0; nt < 2; ++nt) {
      const int gm0 = blockM + m0 + mt * 16 + half * 8;
      const int gn  = blockN + n0 + nt * 16 + lm;
      const float bv = BIAS ? p.bias[gn] : 0.0f;
      const long long cOff = cBase + (long long)gm0 * p.ldc + gn;
      const long long aOff =
          ADDC ? ((long long)g * p.sAdd + (long long)gm0 * p.N + gn) : 0;
#pragma unroll
      for (int r = 0; r < 8; ++r) {
        float v = p.alpha * acc[mt][nt][r] + bv;
        if (ADDC) v += p.beta * p.addC[aOff + (long long)r * p.N];
        if (RELU) v = fmaxf(v, 0.0f);
        if (OUTBF) p.Cb[cOff + (long long)r * p.ldc] = f2bf(v);
        else       p.Cf[cOff + (long long)r * p.ldc] = v;
      }
    }
  }
}

// ---------------------------------------------------------------------------
// Elementwise / reduction kernels
// ---------------------------------------------------------------------------
__global__ void conv_bf_kernel(const float* in, unsigned short* out, long long n) {
  long long i = (long long)blockIdx.x * blockDim.x + threadIdx.x;
  long long s = (long long)gridDim.x * blockDim.x;
  for (; i < n; i += s) out[i] = f2bf(in[i]);
}

__global__ void add2_bf_kernel(const float* a, const float* b,
                               unsigned short* out, long long n) {
  long long i = (long long)blockIdx.x * blockDim.x + threadIdx.x;
  long long s = (long long)gridDim.x * blockDim.x;
  for (; i < n; i += s) out[i] = f2bf(a[i] + b[i]);
}

// out[(res*B+b)*MSA+c] = msa_row0[(b*NRES+res)*MSA+c]  (bf16)
__global__ void msa_gather_kernel(const float* msa, unsigned short* out) {
  long long i = (long long)blockIdx.x * 256 + threadIdx.x;  // exact: 2048*256
  int c = (int)(i & (MSA_DIM - 1));
  long long rb  = i >> 8;
  long long res = rb >> 2;
  long long b   = rb & 3;
  out[i] = f2bf(msa[(b * NRES + res) * MSA_DIM + c]);
}

// out[ib][p] = mean_j par[ib][j][p]   (ib = res*B+b), bf16
__global__ void pair_mean_kernel(const float* par, unsigned short* out) {
  int ib = blockIdx.x;          // 2048 blocks
  int pp = threadIdx.x;         // 128 threads
  const float* base = par + ((long long)ib * NRES) * PAIR_DIM + pp;
  float s = 0.0f;
  for (int j = 0; j < NRES; ++j) s += base[(long long)j * PAIR_DIM];
  out[(long long)ib * PAIR_DIM + pp] = f2bf(s * (1.0f / NRES));
}

// y = LN(a+b) * g + be; rows of 512; writes fp32 and optional bf16 copy.
__global__ __launch_bounds__(256) void ln_kernel(const float* a, const float* b,
                                                 const float* gw, const float* bw,
                                                 float* outf, unsigned short* outb) {
  __shared__ float red[256];
  long long row = blockIdx.x;
  int t = threadIdx.x;
  const float* pa = a + row * D_MODEL;
  const float* pb = b + row * D_MODEL;
  float x0 = pa[t] + pb[t];
  float x1 = pa[t + 256] + pb[t + 256];
  red[t] = x0 + x1;
  __syncthreads();
  for (int o = 128; o > 0; o >>= 1) { if (t < o) red[t] += red[t + o]; __syncthreads(); }
  float mean = red[0] * (1.0f / D_MODEL);
  __syncthreads();
  float d0 = x0 - mean, d1 = x1 - mean;
  red[t] = d0 * d0 + d1 * d1;
  __syncthreads();
  for (int o = 128; o > 0; o >>= 1) { if (t < o) red[t] += red[t + o]; __syncthreads(); }
  float rstd = rsqrtf(red[0] * (1.0f / D_MODEL) + 1e-5f);
  float y0 = d0 * rstd * gw[t] + bw[t];
  float y1 = d1 * rstd * gw[t + 256] + bw[t + 256];
  if (outf) { outf[row * D_MODEL + t] = y0; outf[row * D_MODEL + t + 256] = y1; }
  if (outb) { outb[row * D_MODEL + t] = f2bf(y0);
              outb[row * D_MODEL + t + 256] = f2bf(y1); }
}

// Row softmax over S (256 | S, S <= 4096); bf16 probabilities out.
__global__ __launch_bounds__(256) void softmax_kernel(const float* sc,
                                                      unsigned short* pr, int S) {
  __shared__ float red[256];
  long long row = blockIdx.x;
  int t = threadIdx.x;
  const float* p = sc + row * (long long)S;
  unsigned short* o = pr + row * (long long)S;
  int cnt = S >> 8;             // <= 16
  float ev[16];
  float mx = -3.402823e38f;
  for (int j = 0; j < cnt; ++j) mx = fmaxf(mx, p[t + j * 256]);
  red[t] = mx; __syncthreads();
  for (int s = 128; s > 0; s >>= 1) { if (t < s) red[t] = fmaxf(red[t], red[t + s]); __syncthreads(); }
  mx = red[0]; __syncthreads();
  float sum = 0.0f;
  for (int j = 0; j < cnt; ++j) { ev[j] = __expf(p[t + j * 256] - mx); sum += ev[j]; }
  red[t] = sum; __syncthreads();
  for (int s = 128; s > 0; s >>= 1) { if (t < s) red[t] += red[t + s]; __syncthreads(); }
  float inv = 1.0f / red[0];
  for (int j = 0; j < cnt; ++j) o[t + j * 256] = f2bf(ev[j] * inv);
}

// vt[((b*H+h)*64 + d)*S + s] = vproj[(s*B+b)*D + h*64 + d]   (bf16)
__global__ void vtrans_kernel(const unsigned short* vp, unsigned short* vt, int S) {
  long long n = (long long)NBH * DHEAD * S;
  long long i = (long long)blockIdx.x * blockDim.x + threadIdx.x;
  long long st = (long long)gridDim.x * blockDim.x;
  for (; i < n; i += st) {
    long long s = i % S;
    long long r = i / S;
    int d  = (int)(r & 63);
    int gg = (int)(r >> 6);
    int b  = gg >> 3, h = gg & 7;
    vt[i] = vp[(s * BATCH + b) * D_MODEL + h * DHEAD + d];
  }
}

// ---------------------------------------------------------------------------
// Host driver
// ---------------------------------------------------------------------------
static inline void launch_gemm(hipStream_t st, const GemmP& p, int batch) {
  dim3 grid(p.N / BN, p.M / BM, batch);
  dim3 blk(256);
  const bool bias = p.bias != nullptr;
  const bool add  = p.addC != nullptr;
  const bool relu = p.relu != 0;
  const bool obf  = p.Cb != nullptr;
  if      ( bias && !add && !relu && !obf) gemm_bf16_wmma<true , false, false, false><<<grid, blk, 0, st>>>(p);
  else if ( bias && !add && !relu &&  obf) gemm_bf16_wmma<true , false, false, true ><<<grid, blk, 0, st>>>(p);
  else if ( bias && !add &&  relu &&  obf) gemm_bf16_wmma<true , false, true , true ><<<grid, blk, 0, st>>>(p);
  else if (!bias && !add && !relu && !obf) gemm_bf16_wmma<false, false, false, false><<<grid, blk, 0, st>>>(p);
  else if (!bias &&  add && !relu && !obf) gemm_bf16_wmma<false, true , false, false><<<grid, blk, 0, st>>>(p);
  else if (!bias && !add && !relu &&  obf) gemm_bf16_wmma<false, false, false, true ><<<grid, blk, 0, st>>>(p);
  else                                     gemm_bf16_wmma<true , true , true , true ><<<grid, blk, 0, st>>>(p);
}

extern "C" void kernel_launch(void* const* d_in, const int* in_sizes, int n_in,
                              void* d_out, int out_size, void* d_ws, size_t ws_size,
                              hipStream_t stream) {
  (void)in_sizes; (void)n_in; (void)out_size;

  const float* tgt_sgl  = (const float*)d_in[0];
  const float* tgt_msa  = (const float*)d_in[1];
  const float* tgt_par  = (const float*)d_in[2];
  const float* aa_embed = (const float*)d_in[3];
  const float* dens     = (const float*)d_in[4];
  /* d_in[5] density_mask: unused by reference */
  const float* den_pos  = (const float*)d_in[6];
  const float* den_wei  = (const float*)d_in[7];
  const float* W_ms = (const float*)d_in[8];   const float* b_ms = (const float*)d_in[9];
  const float* W_ps = (const float*)d_in[10];  const float* b_ps = (const float*)d_in[11];
  const float* g_ms = (const float*)d_in[12];  const float* be_ms = (const float*)d_in[13];
  const float* g_ps = (const float*)d_in[14];  const float* be_ps = (const float*)d_in[15];
  const float* g0   = (const float*)d_in[16];  const float* be0 = (const float*)d_in[17];
  const float* sa_Wqkv = (const float*)d_in[18]; const float* sa_bqkv = (const float*)d_in[19];
  const float* sa_Wo   = (const float*)d_in[20]; const float* sa_bo   = (const float*)d_in[21];
  const float* g1   = (const float*)d_in[22];  const float* be1 = (const float*)d_in[23];
  const float* ca_Wqkv = (const float*)d_in[24]; const float* ca_bqkv = (const float*)d_in[25];
  const float* ca_Wo   = (const float*)d_in[26]; const float* ca_bo   = (const float*)d_in[27];
  const float* g2   = (const float*)d_in[28];  const float* be2 = (const float*)d_in[29];
  const float* W1   = (const float*)d_in[30];  const float* b1  = (const float*)d_in[31];
  const float* W2   = (const float*)d_in[32];  const float* b2  = (const float*)d_in[33];
  const float* g3   = (const float*)d_in[34];  const float* be3 = (const float*)d_in[35];

  // Bump allocator over d_ws (256B aligned).
  size_t off = 0;
  auto alloc = [&](size_t bytes) -> void* {
    size_t a = (off + 255) & ~(size_t)255;
    off = a + bytes;
    return (void*)((char*)d_ws + a);
  };
  auto abf = [&](size_t n) { return (unsigned short*)alloc(n * 2); };
  auto af32 = [&](size_t n) { return (float*)alloc(n * 4); };

  // bf16 weights
  unsigned short* wWms   = abf((size_t)D_MODEL * MSA_DIM);
  unsigned short* wWps   = abf((size_t)D_MODEL * PAIR_DIM);
  unsigned short* wSaQkv = abf((size_t)3 * D_MODEL * D_MODEL);
  unsigned short* wSaWo  = abf((size_t)D_MODEL * D_MODEL);
  unsigned short* wCaQkv = abf((size_t)3 * D_MODEL * D_MODEL);
  unsigned short* wCaWo  = abf((size_t)D_MODEL * D_MODEL);
  unsigned short* wW1    = abf((size_t)FFDIM * D_MODEL);
  unsigned short* wW2    = abf((size_t)D_MODEL * FFDIM);
  // bf16 activations
  unsigned short* msA    = abf((size_t)MROWS * MSA_DIM);
  unsigned short* psA    = abf((size_t)MROWS * PAIR_DIM);
  unsigned short* x0b    = abf((size_t)MROWS * D_MODEL);
  unsigned short* qkb    = abf((size_t)MROWS * D_MODEL);
  unsigned short* saq    = abf((size_t)MROWS * D_MODEL);
  unsigned short* sak    = abf((size_t)MROWS * D_MODEL);
  unsigned short* sav    = abf((size_t)MROWS * D_MODEL);
  unsigned short* vtsa   = abf((size_t)NBH * DHEAD * NRES);
  unsigned short* prsa   = abf((size_t)NBH * NRES * NRES);
  unsigned short* attnm  = abf((size_t)MROWS * D_MODEL);
  unsigned short* x1b    = abf((size_t)MROWS * D_MODEL);
  unsigned short* qryb   = abf((size_t)MROWS * D_MODEL);
  unsigned short* keymb  = abf((size_t)DROWS * D_MODEL);
  unsigned short* densb  = abf((size_t)DROWS * D_MODEL);
  unsigned short* caq    = abf((size_t)MROWS * D_MODEL);
  unsigned short* cak    = abf((size_t)DROWS * D_MODEL);
  unsigned short* cav    = abf((size_t)DROWS * D_MODEL);
  unsigned short* vtca   = abf((size_t)NBH * DHEAD * NDEN);
  unsigned short* prca   = abf((size_t)NBH * NRES * NDEN);
  unsigned short* attnm2 = abf((size_t)MROWS * D_MODEL);
  unsigned short* x2b    = abf((size_t)MROWS * D_MODEL);
  unsigned short* ffn1   = abf((size_t)MROWS * FFDIM);
  // fp32 buffers
  float* ms   = af32((size_t)MROWS * D_MODEL);
  float* ps   = af32((size_t)MROWS * D_MODEL);
  float* xms  = af32((size_t)MROWS * D_MODEL);
  float* xps  = af32((size_t)MROWS * D_MODEL);
  float* x0   = af32((size_t)MROWS * D_MODEL);
  float* sao  = af32((size_t)MROWS * D_MODEL);
  float* x1   = af32((size_t)MROWS * D_MODEL);
  float* cao  = af32((size_t)MROWS * D_MODEL);
  float* x2   = af32((size_t)MROWS * D_MODEL);
  float* ffn2 = af32((size_t)MROWS * D_MODEL);
  float* ssa  = af32((size_t)NBH * NRES * NRES);
  float* sca  = af32((size_t)NBH * NRES * NDEN);

  if (off > ws_size) return;  // workspace too small; nothing safe to do

  auto conv = [&](const float* src, unsigned short* dst, long long n) {
    int blocks = (int)((n + 255) / 256); if (blocks > 4096) blocks = 4096;
    conv_bf_kernel<<<dim3(blocks), dim3(256), 0, stream>>>(src, dst, n);
  };
  auto add2 = [&](const float* a, const float* b, unsigned short* dst, long long n) {
    int blocks = (int)((n + 255) / 256); if (blocks > 4096) blocks = 4096;
    add2_bf_kernel<<<dim3(blocks), dim3(256), 0, stream>>>(a, b, dst, n);
  };
  auto G = [&](const unsigned short* A, const unsigned short* B, float* Cf,
               unsigned short* Cb, const float* bias, int M, int N, int K) {
    GemmP p{};
    p.A = A; p.B = B; p.Cf = Cf; p.Cb = Cb; p.bias = bias; p.addC = nullptr;
    p.sA = p.sB = p.sC = p.sAdd = 0;
    p.lda = K; p.ldb = K; p.ldc = N;
    p.M = M; p.N = N; p.K = K;
    p.alpha = 1.0f; p.beta = 0.0f; p.relu = 0;
    p.amode = p.bmode = p.cmode = 0;
    return p;
  };

  // ---- weight conversions ----
  conv(W_ms, wWms, (long long)D_MODEL * MSA_DIM);
  conv(W_ps, wWps, (long long)D_MODEL * PAIR_DIM);
  conv(sa_Wqkv, wSaQkv, 3LL * D_MODEL * D_MODEL);
  conv(sa_Wo, wSaWo, (long long)D_MODEL * D_MODEL);
  conv(ca_Wqkv, wCaQkv, 3LL * D_MODEL * D_MODEL);
  conv(ca_Wo, wCaWo, (long long)D_MODEL * D_MODEL);
  conv(W1, wW1, (long long)FFDIM * D_MODEL);
  conv(W2, wW2, (long long)D_MODEL * FFDIM);

  // ---- MSA gather + pair mean ----
  msa_gather_kernel<<<dim3(MROWS), dim3(256), 0, stream>>>(tgt_msa, msA);
  pair_mean_kernel<<<dim3(MROWS), dim3(PAIR_DIM), 0, stream>>>(tgt_par, psA);

  // ms = msA @ W_ms^T + b_ms ; ps = psA @ W_ps^T + b_ps
  launch_gemm(stream, G(msA, wWms, ms, nullptr, b_ms, MROWS, D_MODEL, MSA_DIM), 1);
  launch_gemm(stream, G(psA, wWps, ps, nullptr, b_ps, MROWS, D_MODEL, PAIR_DIM), 1);

  // x_ms, x_ps, x0
  ln_kernel<<<dim3(MROWS), dim3(256), 0, stream>>>(tgt_sgl, ms, g_ms, be_ms, xms, nullptr);
  ln_kernel<<<dim3(MROWS), dim3(256), 0, stream>>>(tgt_sgl, ps, g_ps, be_ps, xps, nullptr);
  ln_kernel<<<dim3(MROWS), dim3(256), 0, stream>>>(xms, xps, g0, be0, x0, x0b);

  // ---- self attention ----
  add2(x0, aa_embed, qkb, (long long)MROWS * D_MODEL);
  launch_gemm(stream, G(qkb, wSaQkv, nullptr, saq, sa_bqkv, MROWS, D_MODEL, D_MODEL), 1);
  launch_gemm(stream, G(qkb, wSaQkv + (size_t)D_MODEL * D_MODEL, nullptr, sak,
                        sa_bqkv + D_MODEL, MROWS, D_MODEL, D_MODEL), 1);
  launch_gemm(stream, G(x0b, wSaQkv + 2 * (size_t)D_MODEL * D_MODEL, nullptr, sav,
                        sa_bqkv + 2 * D_MODEL, MROWS, D_MODEL, D_MODEL), 1);
  vtrans_kernel<<<dim3(2048), dim3(256), 0, stream>>>(sav, vtsa, NRES);

  {  // scores = 0.125 * Q K^T   (batched over 32 heads, head-merge A/B)
    GemmP p = G(saq, sak, ssa, nullptr, nullptr, NRES, NRES, DHEAD);
    p.amode = 1; p.bmode = 1; p.lda = BATCH * D_MODEL; p.ldb = BATCH * D_MODEL;
    p.alpha = 0.125f; p.sC = (long long)NRES * NRES;
    launch_gemm(stream, p, NBH);
  }
  softmax_kernel<<<dim3(NBH * NRES), dim3(256), 0, stream>>>(ssa, prsa, NRES);
  {  // context = P V  -> merged [L*B, D]
    GemmP p = G(prsa, vtsa, nullptr, attnm, nullptr, NRES, DHEAD, NRES);
    p.sA = (long long)NRES * NRES; p.sB = (long long)DHEAD * NRES;
    p.cmode = 1; p.ldc = BATCH * D_MODEL;
    launch_gemm(stream, p, NBH);
  }
  launch_gemm(stream, G(attnm, wSaWo, sao, nullptr, sa_bo, MROWS, D_MODEL, D_MODEL), 1);
  ln_kernel<<<dim3(MROWS), dim3(256), 0, stream>>>(x0, sao, g1, be1, x1, x1b);

  // ---- cross attention into density memory ----
  add2(x1, aa_embed, qryb, (long long)MROWS * D_MODEL);
  add2(dens, den_pos, keymb, (long long)DROWS * D_MODEL);
  conv(dens, densb, (long long)DROWS * D_MODEL);
  launch_gemm(stream, G(qryb, wCaQkv, nullptr, caq, ca_bqkv, MROWS, D_MODEL, D_MODEL), 1);
  launch_gemm(stream, G(keymb, wCaQkv + (size_t)D_MODEL * D_MODEL, nullptr, cak,
                        ca_bqkv + D_MODEL, DROWS, D_MODEL, D_MODEL), 1);
  launch_gemm(stream, G(densb, wCaQkv + 2 * (size_t)D_MODEL * D_MODEL, nullptr, cav,
                        ca_bqkv + 2 * D_MODEL, DROWS, D_MODEL, D_MODEL), 1);
  vtrans_kernel<<<dim3(4096), dim3(256), 0, stream>>>(cav, vtca, NDEN);

  {  // scores = 0.125 * Q K^T + 8 * den_wei
    GemmP p = G(caq, cak, sca, nullptr, nullptr, NRES, NDEN, DHEAD);
    p.amode = 1; p.bmode = 1; p.lda = BATCH * D_MODEL; p.ldb = BATCH * D_MODEL;
    p.alpha = 0.125f;
    p.addC = den_wei; p.beta = 8.0f; p.sAdd = (long long)NRES * NDEN;
    p.sC = (long long)NRES * NDEN;
    launch_gemm(stream, p, NBH);
  }
  softmax_kernel<<<dim3(NBH * NRES), dim3(256), 0, stream>>>(sca, prca, NDEN);
  {
    GemmP p = G(prca, vtca, nullptr, attnm2, nullptr, NRES, DHEAD, NDEN);
    p.sA = (long long)NRES * NDEN; p.sB = (long long)DHEAD * NDEN;
    p.cmode = 1; p.ldc = BATCH * D_MODEL;
    launch_gemm(stream, p, NBH);
  }
  launch_gemm(stream, G(attnm2, wCaWo, cao, nullptr, ca_bo, MROWS, D_MODEL, D_MODEL), 1);
  ln_kernel<<<dim3(MROWS), dim3(256), 0, stream>>>(x1, cao, g2, be2, x2, x2b);

  // ---- FFN ----
  {
    GemmP p = G(x2b, wW1, nullptr, ffn1, b1, MROWS, FFDIM, D_MODEL);
    p.relu = 1;
    launch_gemm(stream, p, 1);
  }
  launch_gemm(stream, G(ffn1, wW2, ffn2, nullptr, b2, MROWS, D_MODEL, FFDIM), 1);

  // ---- final LN -> output (fp32) ----
  ln_kernel<<<dim3(MROWS), dim3(256), 0, stream>>>(x2, ffn2, g3, be3,
                                                   (float*)d_out, nullptr);
}
